// DeepSeekMoE_76244259439337
// MI455X (gfx1250) — matrix operand order repaired and checked by reading, not compiled, not run on hardware
//
#include <hip/hip_runtime.h>
#include <hip/hip_bf16.h>
#include <stdint.h>

#define NUM_E   8
#define TOPK    2
#define D_MODEL 1024
#define F_FFN   4096

#define MT 128   // M tile
#define NT 128   // N tile
#define KC 64    // K chunk staged in LDS
#define KP 72    // padded LDS row stride (elements); 144B = 16B-aligned rows

typedef __bf16 bf16_t;
typedef bf16_t v16bf __attribute__((ext_vector_type(16)));
typedef float  v8f   __attribute__((ext_vector_type(8)));

__device__ __forceinline__ unsigned short f2bf(float f) {
  unsigned u = __builtin_bit_cast(unsigned, f);
  u += 0x7FFFu + ((u >> 16) & 1u);           // round-to-nearest-even
  return (unsigned short)(u >> 16);
}

// CDNA5 async global->LDS copy (ASYNCcnt-tracked), 16B per lane.
__device__ __forceinline__ void async_copy_b128(unsigned lds_off, const void* gptr) {
  unsigned long long ga = (unsigned long long)(uintptr_t)gptr;
  asm volatile("global_load_async_to_lds_b128 %0, %1, off"
               :: "v"(lds_off), "v"(ga)
               : "memory");
}
__device__ __forceinline__ void wait_asynccnt0() {
  asm volatile("s_wait_asynccnt 0x0" ::: "memory");
}

// ---------------------------------------------------------------- zero init
__global__ void moe_zero_kernel(float* __restrict__ out, size_t n, int* __restrict__ counts) {
  size_t i = (size_t)blockIdx.x * blockDim.x + threadIdx.x;
  if (i < n) out[i] = 0.0f;
  if (i < NUM_E) counts[i] = 0;
}

// ---------------------------------------------------------------- gating
__global__ void moe_gate_kernel(const float* __restrict__ x, const float* __restrict__ gw,
                                const float* __restrict__ gb, int T,
                                int* __restrict__ eids, float* __restrict__ wts,
                                int* __restrict__ counts) {
  int t = blockIdx.x * blockDim.x + threadIdx.x;
  if (t >= T) return;
  float acc[NUM_E];
#pragma unroll
  for (int e = 0; e < NUM_E; ++e) acc[e] = gb[e];
  const float4* xr = (const float4*)(x + (size_t)t * D_MODEL);
  for (int d4 = 0; d4 < D_MODEL / 4; ++d4) {
    float4 xv = xr[d4];
    const float* g = gw + (size_t)d4 * 4 * NUM_E;
#pragma unroll
    for (int j = 0; j < 4; ++j) {
      float xs = (&xv.x)[j];
#pragma unroll
      for (int e = 0; e < NUM_E; ++e) acc[e] = fmaf(xs, g[j * NUM_E + e], acc[e]);
    }
  }
  // top-2, lowest index wins ties (matches jax.lax.top_k)
  float best = -INFINITY, second = -INFINITY;
  int bi = 0, si = 0;
#pragma unroll
  for (int e = 0; e < NUM_E; ++e) {
    float v = acc[e];
    if (v > best)        { second = best; si = bi; best = v; bi = e; }
    else if (v > second) { second = v; si = e; }
  }
  float tex = __expf(second - best);
  float w0 = 1.0f / (1.0f + tex);
  float w1 = tex / (1.0f + tex);
  eids[t * 2 + 0] = bi;  eids[t * 2 + 1] = si;
  wts [t * 2 + 0] = w0;  wts [t * 2 + 1] = w1;
  atomicAdd(&counts[bi], 1);
  atomicAdd(&counts[si], 1);
}

// ---------------------------------------------------------------- scan (tiny)
__global__ void moe_scan_kernel(const int* __restrict__ counts, int* __restrict__ offsets,
                                int* __restrict__ cursor) {
  if (threadIdx.x == 0) {
    int off = 0;
    for (int e = 0; e < NUM_E; ++e) { offsets[e] = off; off += counts[e]; cursor[e] = 0; }
  }
}

// ---------------------------------------------------------------- scatter to buckets
__global__ void moe_scatter_kernel(const int* __restrict__ eids, const float* __restrict__ wts,
                                   const int* __restrict__ offsets, int* __restrict__ cursor,
                                   int* __restrict__ row_token, float* __restrict__ row_wt,
                                   int nslots) {
  int s = blockIdx.x * blockDim.x + threadIdx.x;
  if (s >= nslots) return;
  int e = eids[s];
  int pos = offsets[e] + atomicAdd(&cursor[e], 1);
  row_token[pos] = s >> 1;
  row_wt[pos] = wts[s];
}

// ---------------------------------------------------------------- f32 -> bf16 copy
__global__ void moe_cvt_kernel(const float* __restrict__ in, unsigned short* __restrict__ out,
                               size_t n) {
  size_t i = (size_t)blockIdx.x * blockDim.x + threadIdx.x;
  if (i < n) out[i] = f2bf(in[i]);
}

// ---------------- f32 [E][R][C] -> bf16 [E][C][R] tiled transpose-convert
__global__ void moe_tcvt_kernel(const float* __restrict__ in, unsigned short* __restrict__ out,
                                int R, int C) {
  __shared__ float tile[32][33];
  int e = blockIdx.z;
  int c0 = blockIdx.x * 32, r0 = blockIdx.y * 32;
  const float* src = in + (size_t)e * R * C;
  unsigned short* dst = out + (size_t)e * R * C;
  int tx = threadIdx.x, ty = threadIdx.y;   // 32 x 8
#pragma unroll
  for (int i = 0; i < 4; ++i)
    tile[ty + 8 * i][tx] = src[(size_t)(r0 + ty + 8 * i) * C + (c0 + tx)];
  __syncthreads();
#pragma unroll
  for (int i = 0; i < 4; ++i)
    dst[(size_t)(c0 + ty + 8 * i) * R + (r0 + tx)] = f2bf(tile[tx][ty + 8 * i]);
}

// ---------------------------------------------------------------- grouped NT GEMM
// PASS 1: H[base+m][:] = silu(Xbf[row_token[m]][:] @ W1T[e]^T + b1[e])   (K=D, N=F)
// PASS 2: out[token][:] += row_wt[m] * (H[base+m][:] @ W2T[e]^T + b2[e]) (K=F, N=D)
template <int PASS>
__global__ __launch_bounds__(256)
void moe_gemm_kernel(const unsigned short* __restrict__ Abase,
                     const unsigned short* __restrict__ Ball,   // [E][N][K] bf16
                     const float* __restrict__ bias,            // [E][N]
                     const int* __restrict__ counts, const int* __restrict__ offsets,
                     const int* __restrict__ row_token, const float* __restrict__ row_wt,
                     unsigned short* __restrict__ Hout, float* __restrict__ Yout,
                     int K, int N) {
  int e = blockIdx.z;
  int cnt = counts[e];
  int m0 = (int)blockIdx.y * MT;
  if (m0 >= cnt) return;
  int base = offsets[e];
  int n0 = (int)blockIdx.x * NT;

  // double-buffered tiles; async global->LDS fills one while WMMA consumes the other
  __shared__ unsigned short As[2][MT * KP];
  __shared__ unsigned short Bs[2][NT * KP];

  int tid = threadIdx.x;
  int lane = tid & 31;
  int wave = tid >> 5;
  int wm = wave >> 2;   // 0..1 -> 64-row strip
  int wn = wave & 3;    // 0..3 -> 32-col strip

  // staging map: each thread owns 4 x 16B segments of A rows and 4 of B rows
  int srow = tid >> 3;  // 0..31
  int sseg = tid & 7;   // 0..7
  const unsigned short* aptr[4];
  const unsigned short* bptr[4];
  const unsigned short* Bsrc = Ball + (size_t)e * (size_t)K * (size_t)N;
#pragma unroll
  for (int i = 0; i < 4; ++i) {
    int ar = srow + 32 * i;
    int grow = m0 + ar;
    if (PASS == 1) {
      int tok = (grow < cnt) ? row_token[base + grow] : row_token[base];
      aptr[i] = Abase + (size_t)tok * (size_t)K;
    } else {
      int rr = (grow < cnt) ? grow : 0;
      aptr[i] = Abase + (size_t)(base + rr) * (size_t)K;
    }
    bptr[i] = Bsrc + (size_t)(n0 + srow + 32 * i) * (size_t)K;
  }

  // issue one stage of async copies into LDS buffer `bufi`
  auto stage = [&](int k0, int bufi) {
#pragma unroll
    for (int i = 0; i < 4; ++i) {
      int r = srow + 32 * i;
      unsigned loff_a = (unsigned)(uintptr_t)&As[bufi][r * KP + sseg * 8];
      unsigned loff_b = (unsigned)(uintptr_t)&Bs[bufi][r * KP + sseg * 8];
      async_copy_b128(loff_a, aptr[i] + k0 + sseg * 8);
      async_copy_b128(loff_b, bptr[i] + k0 + sseg * 8);
    }
  };

  v8f acc[4][2];
#pragma unroll
  for (int mi = 0; mi < 4; ++mi)
#pragma unroll
    for (int ni = 0; ni < 2; ++ni) acc[mi][ni] = v8f{};

  int arow = lane & 15;
  int koff = (lane >> 4) * 16;   // A-frag K sub-offset (contiguous k = 16h + s, matches B)
  int kbo  = (lane >> 4) * 16;   // B-frag K sub-offset (contiguous k = 16h + s, matches A)

  int nstages = K / KC;
  stage(0, 0);
  for (int s = 0; s < nstages; ++s) {
    wait_asynccnt0();      // our stage-s copies have landed
    __syncthreads();       // everyone's stage-s copies landed; compute(s-1) fully drained
    if (s + 1 < nstages) stage((s + 1) * KC, (s + 1) & 1);  // overlap with compute(s)

    const unsigned short* Ab = As[s & 1];
    const unsigned short* Bb = Bs[s & 1];
#pragma unroll
    for (int ks = 0; ks < KC; ks += 32) {
      union { v16bf v; uint4 q[2]; } bf[2];
#pragma unroll
      for (int ni = 0; ni < 2; ++ni) {
        const unsigned short* p = &Bb[(wn * 32 + ni * 16 + (lane & 15)) * KP + ks + kbo];
        bf[ni].q[0] = *(const uint4*)(p);
        bf[ni].q[1] = *(const uint4*)(p + 8);
      }
#pragma unroll
      for (int mi = 0; mi < 4; ++mi) {
        union { v16bf v; uint4 q[2]; } af;
        const unsigned short* p = &Ab[(wm * 64 + mi * 16 + arow) * KP + ks + koff];
        af.q[0] = *(const uint4*)(p);
        af.q[1] = *(const uint4*)(p + 8);
#pragma unroll
        for (int ni = 0; ni < 2; ++ni) {
          acc[mi][ni] = __builtin_amdgcn_wmma_f32_16x16x32_bf16(
              false, af.v, false, bf[ni].v, (short)0, acc[mi][ni], false, false);
        }
      }
    }
  }

  // epilogue; C/D layout: lane n = lane&15, VGPR r -> m = r + 8*(lane>>4)
  int nn = lane & 15;
  int mm8 = (lane >> 4) * 8;
#pragma unroll
  for (int mi = 0; mi < 4; ++mi)
#pragma unroll
    for (int ni = 0; ni < 2; ++ni)
#pragma unroll
      for (int r = 0; r < 8; ++r) {
        int m_loc = wm * 64 + mi * 16 + mm8 + r;
        int grow = m0 + m_loc;
        if (grow >= cnt) continue;
        int gn = n0 + wn * 32 + ni * 16 + nn;
        float v = acc[mi][ni][r];
        if (PASS == 1) {
          float h = v + bias[(size_t)e * N + gn];
          h = h / (1.0f + __expf(-h));            // silu
          Hout[(size_t)(base + grow) * N + gn] = f2bf(h);
        } else {
          float w = row_wt[base + grow];
          float y = (v + bias[(size_t)e * N + gn]) * w;
          int tok = row_token[base + grow];
          atomicAdd(&Yout[(size_t)tok * N + gn], y);
        }
      }
}

// ---------------------------------------------------------------- launcher
extern "C" void kernel_launch(void* const* d_in, const int* in_sizes, int n_in,
                              void* d_out, int out_size, void* d_ws, size_t ws_size,
                              hipStream_t stream) {
  const float* x  = (const float*)d_in[0];
  const float* gw = (const float*)d_in[1];
  const float* gb = (const float*)d_in[2];
  const float* W1 = (const float*)d_in[3];
  const float* b1 = (const float*)d_in[4];
  const float* W2 = (const float*)d_in[5];
  const float* b2 = (const float*)d_in[6];
  float* out = (float*)d_out;

  const int D = D_MODEL, F = F_FFN, E = NUM_E;
  const int T = in_sizes[0] / D;   // 8192 tokens
  const int RT = T * TOPK;         // 16384 routed rows
  (void)n_in; (void)out_size; (void)ws_size;

  char* ws = (char*)d_ws;
  size_t off = 0;
  auto alloc = [&](size_t bytes) -> char* {
    char* p = ws + off;
    off += (bytes + 255) & ~(size_t)255;
    return p;
  };
  int*   counts    = (int*)  alloc(E * sizeof(int));
  int*   offsets   = (int*)  alloc(E * sizeof(int));
  int*   cursor    = (int*)  alloc(E * sizeof(int));
  int*   eids      = (int*)  alloc((size_t)RT * sizeof(int));
  float* wts       = (float*)alloc((size_t)RT * sizeof(float));
  int*   row_token = (int*)  alloc((size_t)RT * sizeof(int));
  float* row_wt    = (float*)alloc((size_t)RT * sizeof(float));
  unsigned short* Xbf = (unsigned short*)alloc((size_t)T * D * 2);
  unsigned short* W1T = (unsigned short*)alloc((size_t)E * D * F * 2);
  unsigned short* W2T = (unsigned short*)alloc((size_t)E * D * F * 2);
  unsigned short* H   = (unsigned short*)alloc((size_t)RT * F * 2);

  // 1) zero output + counts
  {
    size_t n = (size_t)T * D;
    moe_zero_kernel<<<(int)((n + 255) / 256), 256, 0, stream>>>(out, n, counts);
  }
  // 2) gate logits + top-2 softmax + expert counts
  moe_gate_kernel<<<(T + 127) / 128, 128, 0, stream>>>(x, gw, gb, T, eids, wts, counts);
  // 3) exclusive scan of counts
  moe_scan_kernel<<<1, 32, 0, stream>>>(counts, offsets, cursor);
  // 4) scatter (token, weight) into per-expert buckets
  moe_scatter_kernel<<<(RT + 255) / 256, 256, 0, stream>>>(eids, wts, offsets, cursor,
                                                           row_token, row_wt, RT);
  // 5) x -> bf16
  {
    size_t n = (size_t)T * D;
    moe_cvt_kernel<<<(int)((n + 255) / 256), 256, 0, stream>>>(x, Xbf, n);
  }
  // 6) W1 [E][D][F] -> bf16 [E][F][D]
  {
    dim3 g(F / 32, D / 32, E), b(32, 8);
    moe_tcvt_kernel<<<g, b, 0, stream>>>(W1, W1T, D, F);
  }
  // 7) W2 [E][F][D] -> bf16 [E][D][F]
  {
    dim3 g(D / 32, F / 32, E), b(32, 8);
    moe_tcvt_kernel<<<g, b, 0, stream>>>(W2, W2T, F, D);
  }
  // 8) pass 1: H = silu(X @ W1 + b1), grouped per expert  (K=D, N=F)
  {
    dim3 g(F / NT, (RT + MT - 1) / MT, E);
    moe_gemm_kernel<1><<<g, 256, 0, stream>>>(Xbf, W1T, b1, counts, offsets,
                                              row_token, row_wt, H, nullptr, D, F);
  }
  // 9) pass 2: out += w * (H @ W2 + b2)  (K=F, N=D); exactly 2 atomic adds/element
  {
    dim3 g(D / NT, (RT + MT - 1) / MT, E);
    moe_gemm_kernel<2><<<g, 256, 0, stream>>>(H, W2T, b2, counts, offsets,
                                              row_token, row_wt, nullptr, out, F, D);
  }
}
